// FusionAVWGCN_6657199308827
// MI455X (gfx1250) — matrix operand order repaired and checked by reading, not compiled, not run on hardware
//
#include <hip/hip_runtime.h>

typedef __attribute__((ext_vector_type(16))) _Float16 v16h;
typedef __attribute__((ext_vector_type(8)))  float    v8f;
typedef __attribute__((ext_vector_type(4)))  float    v4f;

#define N_NODES 8192
#define DIM     16      // node-embed dim
#define C_IN    32
#define C_OUT   32
#define NBATCH  8
#define JTOT    256     // B*C_IN columns of the fused A@X
#define ROWS    32      // output rows per workgroup (2 halves of 16)
#define MT      32      // source-node tile per iteration
#define XROW    40      // halves per XsT LDS row (32 data + 8 pad; 80B, 16B-aligned)

// ---- CDNA5 async global->LDS copies (ASYNCcnt-tracked, §08_async_tensor) ----
__device__ __forceinline__ void async_ld_b128(unsigned lds_off, unsigned long long ga) {
  asm volatile("global_load_async_to_lds_b128 %0, %1, off"
               :: "v"(lds_off), "v"(ga) : "memory");
}
__device__ __forceinline__ void async_ld_b64(unsigned lds_off, unsigned long long ga) {
  asm volatile("global_load_async_to_lds_b64 %0, %1, off"
               :: "v"(lds_off), "v"(ga) : "memory");
}
__device__ __forceinline__ void wait_async_le5() {
  asm volatile("s_wait_asynccnt 0x5" ::: "memory");
}

// ---------------------------------------------------------------------------
// Kernel 0 (prep): XhT[j][m] = (f16) x[b, m, c] with j = b*32+c.
// Column-major f16 copy of Xcat so the attention kernel can DMA contiguous
// 64B rows straight into LDS with async-to-LDS (raw-byte engine => must be
// pre-converted). 4 MB, L2-resident.
// ---------------------------------------------------------------------------
__global__ __launch_bounds__(256) void avwgcn_prep(const float* __restrict__ x,
                                                   _Float16* __restrict__ XhT) {
  const int m = blockIdx.x;          // 0..8191
  const int j = threadIdx.x;         // 0..255  (coalesced read over c)
  const int b = j >> 5, c = j & 31;
  XhT[(size_t)j * N_NODES + m] = (_Float16)x[((size_t)b * N_NODES + m) * C_IN + c];
}

// ---------------------------------------------------------------------------
// Kernel 1: Y[n, b*32+c] = softmax(relu(E E^T))[n,:] @ Xcat[:, b*32+c]
// Flash-attention-style, async-pipelined: each iteration issues exactly 5
// async-to-LDS ops per lane for tile i+1 (double-buffered), waits asynccnt<=5
// (in-order completion => tile i resident), then does WMMA on tile i.
// ---------------------------------------------------------------------------
__global__ __launch_bounds__(256) void avwgcn_attn(const float* __restrict__ E,
                                                   const _Float16* __restrict__ XhT,
                                                   float* __restrict__ Y) {
  __shared__ _Float16 XsT[2][JTOT][XROW];   // 40 KB: X tile, transposed [j][k]
  __shared__ float    Ecs[2][MT][DIM];      //  4 KB: E rows of source tile
  __shared__ float    Ss[2][16][MT + 1];    //  4 KB: relu scores per row-half

  const int tid  = threadIdx.x;
  const int lane = tid & 31;
  const int wv   = tid >> 5;        // wave id 0..7
  const int h    = wv >> 2;         // row half (0/1)
  const int cg   = wv & 3;          // column group (64 cols each)
  const int hh   = lane >> 4;       // lane half (K-half selector)
  const int l16  = lane & 15;
  const int n0   = blockIdx.x * ROWS;
  const int jbase = cg * 64;

  // LDS byte offsets / global bases for this thread's async transfers
  const unsigned xs_off0  = (unsigned)(uintptr_t)&XsT[0][tid][0];
  const unsigned ecs_off0 = (unsigned)(uintptr_t)&Ecs[0][0][0] + tid * 8;
  const unsigned long long gaX0 = (unsigned long long)(uintptr_t)(XhT + (size_t)tid * N_NODES);
  const unsigned long long gaE0 = (unsigned long long)(uintptr_t)E;

  // A-fragment of E rows (16x32 f16, K=16..31 zero-padded) for scorer waves.
  const bool is_scorer = (wv == 0) || (wv == 1) || (wv == 4) || (wv == 5);
  v16h ea;
#pragma unroll
  for (int t = 0; t < 16; ++t) ea[t] = (_Float16)0.f;
  if (is_scorer) {
    const int row = n0 + 16 * h + l16;
#pragma unroll
    for (int t = 0; t < 16; ++t)
      if (hh == 0) ea[t] = (_Float16)E[row * DIM + t];   // K = 16*hh + t (< 16)
  }

  v8f acc[4];
#pragma unroll
  for (int f = 0; f < 4; ++f)
#pragma unroll
    for (int v = 0; v < 8; ++v) acc[f][v] = 0.f;

  float run_max = 0.f;   // relu => true row max >= 0
  float run_sum = 0.f;

  // ---- prologue: async-load tile 0 into buffer 0 (5 issues per lane) ----
#pragma unroll
  for (int t = 0; t < 4; ++t) async_ld_b128(xs_off0 + 16 * t, gaX0 + 16 * t);
  async_ld_b64(ecs_off0, gaE0 + tid * 8);

  for (int m0 = 0; m0 < N_NODES; m0 += MT) {
    const int buf  = (m0 >> 5) & 1;
    const int nbuf = buf ^ 1;
    const int mn   = (m0 + MT) & (N_NODES - 1);        // wrap: dummy reload of tile 0 at end

    // ---- issue tile i+1 (uniform 5 asyncs/lane; target buffer idle since
    //      the barrier that ended iteration i-1) ----
    {
      const unsigned long long gx = gaX0 + (unsigned long long)(2 * mn);
#pragma unroll
      for (int t = 0; t < 4; ++t)
        async_ld_b128(xs_off0 + (unsigned)(nbuf * (int)sizeof(XsT[0])) + 16 * t, gx + 16 * t);
      async_ld_b64(ecs_off0 + (unsigned)(nbuf * (int)sizeof(Ecs[0])),
                   gaE0 + (unsigned long long)(mn * DIM * 4 + tid * 8));
    }
    // oldest 5 (tile i) complete; newest 5 (tile i+1) may remain in flight
    wait_async_le5();
    __syncthreads();

    // ---- scores: S = relu(E_rows @ E_cols^T), one WMMA per 16x16 subtile ----
    if (is_scorer) {
      const int ct = wv & 1;
      const int nn = ct * 16 + l16;
      v16h eb;
      if (hh == 0) {
#pragma unroll
        for (int q = 0; q < 4; ++q) {
          const v4f ev = *(const v4f*)&Ecs[buf][nn][4 * q];
#pragma unroll
          for (int t = 0; t < 4; ++t) eb[4 * q + t] = (_Float16)ev[t];
        }
      } else {
#pragma unroll
        for (int t = 0; t < 16; ++t) eb[t] = (_Float16)0.f;   // K>=16 pad
      }
      v8f s;
#pragma unroll
      for (int v = 0; v < 8; ++v) s[v] = 0.f;
      s = __builtin_amdgcn_wmma_f32_16x16x32_f16(false, ea, false, eb,
                                                 (short)0, s, false, false);
#pragma unroll
      for (int v = 0; v < 8; ++v) {
        const float sv = s[v];
        Ss[h][v + 8 * hh][ct * 16 + l16] = sv > 0.f ? sv : 0.f;
      }
    }
    __syncthreads();

    // ---- online softmax + P@X (all waves) ----
    {
      const int r = l16;
      float tmax = 0.f;
#pragma unroll
      for (int c = 0; c < MT; ++c) tmax = fmaxf(tmax, Ss[h][r][c]);
      const float newmax = fmaxf(run_max, tmax);
      const float sc = __expf(run_max - newmax);
      run_sum *= sc;

      v16h pa;                                   // P as 16x32 f16 A-fragment
#pragma unroll
      for (int t = 0; t < 16; ++t) pa[t] = (_Float16)0.f;
#pragma unroll
      for (int c = 0; c < MT; ++c) {
        const float p = __expf(Ss[h][r][c] - newmax);
        run_sum += p;
        const int cc = c & 15;
        if ((c >> 4) == hh) {
          const int slot = cc;
          pa[slot] = (_Float16)p;
        }
      }
      run_max = newmax;

#pragma unroll
      for (int v = 0; v < 8; ++v) {
        const float srow = __shfl(sc, v + 8 * hh, 32);
#pragma unroll
        for (int f = 0; f < 4; ++f) acc[f][v] *= srow;
      }

      // P (16x32) @ X (32x16): B-frag = 32 contiguous bytes of XsT row
#pragma unroll
      for (int f = 0; f < 4; ++f) {
        const int jn = jbase + 16 * f + l16;
        const v16h xb = *(const v16h*)&XsT[buf][jn][16 * hh];
        acc[f] = __builtin_amdgcn_wmma_f32_16x16x32_f16(false, pa, false, xb,
                                                        (short)0, acc[f],
                                                        false, false);
      }
    }
    __syncthreads();   // protect buf before it becomes the issue target again
  }

  // ---- finalize: divide by softmax denominator, store Y ----
#pragma unroll
  for (int v = 0; v < 8; ++v) {
    const float rs  = __shfl(run_sum, v + 8 * hh, 32);
    const float inv = 1.f / rs;
    const int row   = n0 + 16 * h + v + 8 * hh;
#pragma unroll
    for (int f = 0; f < 4; ++f)
      Y[row * JTOT + jbase + 16 * f + l16] = acc[f][v] * inv;
  }
}

// ---------------------------------------------------------------------------
// Kernel 2 (light, L2-resident): W[n,k,i,o] = sum_d E[n,d] WP[d,k,i,o];
//   out[b,n,o] = bias + sum_i x[b,n,i]*W[n,0,i,o] + sum_i Y[n,b*32+i]*W[n,1,i,o]
// ---------------------------------------------------------------------------
__global__ __launch_bounds__(256) void avwgcn_combine(const float* __restrict__ x,
                                                      const float* __restrict__ E,
                                                      const float* __restrict__ WP,
                                                      const float* __restrict__ BP,
                                                      const float* __restrict__ Y,
                                                      float* __restrict__ out) {
  const int tid = blockIdx.x * blockDim.x + threadIdx.x;
  const int o = tid & 31;
  const int n = tid >> 5;
  if (n >= N_NODES) return;

  float e[DIM];
#pragma unroll
  for (int d = 0; d < DIM; ++d) e[d] = E[n * DIM + d];

  float w[2 * C_IN];
#pragma unroll 2
  for (int k = 0; k < 2; ++k)
#pragma unroll 8
    for (int i = 0; i < C_IN; ++i) {
      float s = 0.f;
#pragma unroll
      for (int d = 0; d < DIM; ++d)
        s += e[d] * WP[((d * 2 + k) * C_IN + i) * C_OUT + o];
      w[k * C_IN + i] = s;
    }

  float bias = 0.f;
#pragma unroll
  for (int d = 0; d < DIM; ++d) bias += e[d] * BP[d * C_OUT + o];

#pragma unroll 2
  for (int b = 0; b < NBATCH; ++b) {
    const float* xr = x + ((size_t)b * N_NODES + n) * C_IN;
    const float* yr = Y + (size_t)n * JTOT + b * C_IN;
    float a = bias;
#pragma unroll 8
    for (int i = 0; i < C_IN; ++i) a += xr[i] * w[i];
#pragma unroll 8
    for (int i = 0; i < C_IN; ++i) a += yr[i] * w[C_IN + i];
    out[((size_t)b * N_NODES + n) * C_OUT + o] = a;
  }
}

extern "C" void kernel_launch(void* const* d_in, const int* in_sizes, int n_in,
                              void* d_out, int out_size, void* d_ws, size_t ws_size,
                              hipStream_t stream) {
  const float* x  = (const float*)d_in[0];   // (8, 8192, 32)
  const float* E  = (const float*)d_in[1];   // (8192, 16)
  const float* WP = (const float*)d_in[2];   // (16, 2, 32, 32)
  const float* BP = (const float*)d_in[3];   // (16, 32)

  float*     Y   = (float*)d_ws;                                   // 8 MB
  _Float16*  XhT = (_Float16*)((char*)d_ws + (size_t)N_NODES * JTOT * 4); // +4 MB
  float*     out = (float*)d_out;            // (8, 8192, 32)

  avwgcn_prep<<<dim3(N_NODES), dim3(JTOT), 0, stream>>>(x, XhT);
  avwgcn_attn<<<dim3(N_NODES / ROWS), dim3(256), 0, stream>>>(E, XhT, Y);
  avwgcn_combine<<<dim3((N_NODES * C_OUT) / 256), dim3(256), 0, stream>>>(
      x, E, WP, BP, Y, out);
}
